// Attention_48541720379807
// MI455X (gfx1250) — matrix operand order repaired and checked by reading, not compiled, not run on hardware
//
#include <hip/hip_runtime.h>
#include <hip/hip_bf16.h>
#include <math.h>

// MI455X (gfx1250) wave32 WMMA attention pipeline.
// Pipeline: pack weights fp16 -> GEMM(k@Wk), GEMM(q@Wq) -> flash attention -> GEMM(out@P^T)+bias.
// All matmuls via v_wmma_f32_16x16x32_f16 (f32 accum). Softmax in fp32.
// fp16 global->LDS tile copies use the Tensor Data Mover (TENSOR_LOAD_TO_LDS,
// TENSORcnt) when the builtin exists, else GLOBAL_LOAD_ASYNC_TO_LDS_B128
// (ASYNCcnt). GEMM k-loop is double-buffered so DMA overlaps WMMA.
// Workspace layout (needs ~102MB of d_ws):
//   [0,2MB)   WKt fp16 (N*H, E)
//   [2,4MB)   WQt fp16 (N*H, E)
//   [4,6MB)   Pt  fp16 (E, N*H)   (proj_w cast; already B^T layout)
//   [6,38MB)  KXh fp16 (B*L, N*H)
//   [38,70MB) QXh fp16 (B*L, N*H)
//   [70,102MB)Oh  fp16 (B*L, N*H)

typedef _Float16 v16h __attribute__((ext_vector_type(16)));
typedef _Float16 v8h  __attribute__((ext_vector_type(8)));
typedef float    v8f  __attribute__((ext_vector_type(8)));
typedef unsigned int u32x4 __attribute__((ext_vector_type(4)));
typedef int          i32x8 __attribute__((ext_vector_type(8)));
typedef int          i32x4 __attribute__((ext_vector_type(4)));

#if __has_builtin(__builtin_amdgcn_tensor_load_to_lds)
#define HAVE_TDM 1
#else
#define HAVE_TDM 0
#endif

static __device__ __forceinline__ v16h cat8(v8h lo, v8h hi) {
    return __builtin_shufflevector(lo, hi, 0,1,2,3,4,5,6,7,8,9,10,11,12,13,14,15);
}

#define WMMA16(A,B,C) __builtin_amdgcn_wmma_f32_16x16x32_f16(false,(A),false,(B),(short)0,(C),false,false)

// Async 16B global->LDS copy (GLOBAL_LOAD_ASYNC_TO_LDS_B128, ASYNCcnt).
// Low 32 bits of a generic shared pointer are the wave-relative LDS address.
static __device__ __forceinline__ void async_copy16(void* lds, const _Float16* g) {
    unsigned ldsoff = (unsigned)(uintptr_t)lds;
    asm volatile("global_load_async_to_lds_b128 %0, %1, off"
                 :: "v"(ldsoff), "v"(g) : "memory");
}

#if HAVE_TDM
// TDM: load a (rows x 32-half) fp16 tile, global row stride = strideElems halves,
// into LDS with pitch 40 halves (pad_amount=4 dwords every pad_interval=16 dwords).
// Per-wave DMA; EXEC ignored; tracked by TENSORcnt. Issue from one wave only.
static __device__ __forceinline__ void tdm_load_tile32(void* lds, const _Float16* g,
                                                       int rows, int strideElems) {
    unsigned long long ga = (unsigned long long)(uintptr_t)g;
    unsigned ldsoff = (unsigned)(uintptr_t)lds;
    u32x4 g0;
    g0[0] = 1u;                                   // count=1 (valid user D#)
    g0[1] = ldsoff;                               // lds_addr (bytes)
    g0[2] = (unsigned)ga;                         // global_addr[31:0]
    g0[3] = (unsigned)((ga >> 32) & 0x01FFFFFFu)  // global_addr[56:32]
          | (0x2u << 30);                         // type = 2 ("image")
    i32x8 g1;
    g1[0] = (1 << 16)      // data_size = 2 bytes
          | (1 << 20)      // pad_enable
          | (3 << 22)      // pad_interval: 16 dwords (one 32-half row)
          | (3 << 25);     // pad_amount: 4 dwords (8 halves) -> pitch 40
    g1[1] = 32 << 16;                 // tensor_dim0 = 32 (bits 63:48)
    g1[2] = (rows & 0xFFFF) << 16;    // tensor_dim0 hi=0 | tensor_dim1 lo
    g1[3] = 32 << 16;                 // tensor_dim1 hi=0 | tile_dim0 = 32
    g1[4] = rows & 0xFFFF;            // tile_dim1 = rows, tile_dim2 = 0
    g1[5] = strideElems;              // tensor_dim0_stride lo32
    g1[6] = 0;
    g1[7] = 0;
    i32x4 zz4 = {0, 0, 0, 0};
    i32x8 zz8 = {0, 0, 0, 0, 0, 0, 0, 0};
    // 6-arg toolchain form: (g0, g1, g2, g3, g4, cpol)
    __builtin_amdgcn_tensor_load_to_lds(g0, g1, zz4, zz4, zz8, 0);
}
#endif

// ---------------- weight packing ----------------
// w: (N,E,H) fp32 -> wt: (N*H, E) fp16, wt[(n*H+h)*E+e] = w[(n*E+e)*H+h]
__global__ void prep_wt(const float* __restrict__ w, _Float16* __restrict__ wt,
                        int E, int H) {
    int idx = blockIdx.x * 256 + threadIdx.x;
    int h = idx % H;
    int e = (idx / H) % E;
    int n = idx / (H * E);
    wt[((size_t)(n * H + h)) * E + e] = (_Float16)w[idx];
}

__global__ void prep_cast(const float* __restrict__ in, _Float16* __restrict__ out) {
    int i = blockIdx.x * 256 + threadIdx.x;
    out[i] = (_Float16)in[i];
}

// ---------------- tiled WMMA GEMM (double buffered) ----------------
// C(M,Nc) = A(M,K) * B(K,Nc), with B given transposed: BT(Nc,K) fp16 row-major.
// Block tile 128x128, 8 waves (4x2), each wave 32x64 (2x4 wmma accs), K step 32.
template <bool A_HALF, bool OUT_FLOAT>
__global__ __launch_bounds__(256) void gemm128(const void* __restrict__ Ap,
                                               const _Float16* __restrict__ BT,
                                               void* __restrict__ Cp,
                                               const float* __restrict__ bias,
                                               int M, int Nc, int K) {
    __shared__ _Float16 As[2][128][40];
    __shared__ _Float16 Bs[2][128][40];

    const int t    = threadIdx.x;
    const int lane = t & 31;
    const int wave = t >> 5;
    const int wr   = (wave >> 1) * 32;
    const int wc   = (wave & 1) * 64;
    const int row0 = blockIdx.y * 128;
    const int col0 = blockIdx.x * 128;
    const int lr   = t >> 3;             // fp32 A loader row (x4 passes)
    const int lc   = (t & 7) * 4;        // fp32 A loader k offset
    const int r2   = t >> 1;             // async fallback row
    const int c2   = (t & 1) * 16;       // async fallback k offset
    const int l15  = lane & 15;
    const int kb8  = (lane >> 4) * 8;
    const int kb16 = (lane >> 4) * 16;

    v8f acc[2][4];
#pragma unroll
    for (int i = 0; i < 2; i++)
#pragma unroll
        for (int j = 0; j < 4; j++)
#pragma unroll
            for (int e = 0; e < 8; e++) acc[i][j][e] = 0.0f;

    float4 areg[4];                      // fp32 A staging (issue->commit split)

    // ---- tile fill: issue phase (no waits) ----
    auto issue = [&](int buf, int k0) {
#if HAVE_TDM
        if (wave == 0) {
            tdm_load_tile32(&Bs[buf][0][0], BT + (size_t)col0 * K + k0, 128, K);
            if (A_HALF)
                tdm_load_tile32(&As[buf][0][0],
                                ((const _Float16*)Ap) + (size_t)row0 * K + k0, 128, K);
        }
#else
        async_copy16(&Bs[buf][r2][c2],     BT + (size_t)(col0 + r2) * K + k0 + c2);
        async_copy16(&Bs[buf][r2][c2 + 8], BT + (size_t)(col0 + r2) * K + k0 + c2 + 8);
        if (A_HALF) {
            const _Float16* A = (const _Float16*)Ap;
            async_copy16(&As[buf][r2][c2],     A + (size_t)(row0 + r2) * K + k0 + c2);
            async_copy16(&As[buf][r2][c2 + 8], A + (size_t)(row0 + r2) * K + k0 + c2 + 8);
        }
#endif
        if (!A_HALF) {
#pragma unroll
            for (int p = 0; p < 4; p++)
                areg[p] = *(const float4*)(((const float*)Ap) +
                                           (size_t)(row0 + p * 32 + lr) * K + k0 + lc);
        }
    };
    // ---- tile fill: commit phase (fp32 convert -> LDS) ----
    auto commit = [&](int buf) {
        if (!A_HALF) {
#pragma unroll
            for (int p = 0; p < 4; p++) {
                const int r = p * 32 + lr;
                As[buf][r][lc + 0] = (_Float16)areg[p].x;
                As[buf][r][lc + 1] = (_Float16)areg[p].y;
                As[buf][r][lc + 2] = (_Float16)areg[p].z;
                As[buf][r][lc + 3] = (_Float16)areg[p].w;
            }
        }
    };

    const int nt = K / 32;
    issue(0, 0);
    commit(0);
    for (int it = 0; it < nt; ++it) {
        const int cur = it & 1;
        const bool have_next = (it + 1) < nt;
        if (have_next) issue(cur ^ 1, (it + 1) * 32);
        // retire current tile's DMA (per-wave counters are in-order)
#if HAVE_TDM
        if (wave == 0) {
            if (have_next) __builtin_amdgcn_s_wait_tensorcnt((short)(A_HALF ? 2 : 1));
            else           __builtin_amdgcn_s_wait_tensorcnt((short)0);
        }
#else
        if (have_next) {
            if (A_HALF) asm volatile("s_wait_asynccnt 0x4" ::: "memory");
            else        asm volatile("s_wait_asynccnt 0x2" ::: "memory");
        } else {
            asm volatile("s_wait_asynccnt 0x0" ::: "memory");
        }
#endif
        __syncthreads();

        v16h af[2];
#pragma unroll
        for (int i = 0; i < 2; i++) {
            const int r = wr + i * 16 + l15;
            v8h lo = *(const v8h*)&As[cur][r][kb8];
            v8h hi = *(const v8h*)&As[cur][r][16 + kb8];
            af[i] = cat8(lo, hi);
        }
        v16h bf[4];
#pragma unroll
        for (int j = 0; j < 4; j++) {
            const int c = wc + j * 16 + l15;
            v8h lo = *(const v8h*)&Bs[cur][c][kb8];
            v8h hi = *(const v8h*)&Bs[cur][c][16 + kb8];
            bf[j] = cat8(lo, hi);
        }
#pragma unroll
        for (int i = 0; i < 2; i++)
#pragma unroll
            for (int j = 0; j < 4; j++)
                acc[i][j] = WMMA16(af[i], bf[j], acc[i][j]);

        if (have_next) commit(cur ^ 1);  // fp32 convert lands after the WMMAs
        __syncthreads();
    }

#pragma unroll
    for (int i = 0; i < 2; i++)
#pragma unroll
        for (int j = 0; j < 4; j++) {
            const int gcol = col0 + wc + j * 16 + l15;
#pragma unroll
            for (int e = 0; e < 8; e++) {
                const int grow = row0 + wr + i * 16 + (lane >> 4) * 8 + e;
                if (OUT_FLOAT)
                    ((float*)Cp)[(size_t)grow * Nc + gcol] = acc[i][j][e] + bias[gcol];
                else
                    ((_Float16*)Cp)[(size_t)grow * Nc + gcol] = (_Float16)acc[i][j][e];
            }
        }
}

// ---------------- flash attention ----------------
// grid: (L/128, N*B). Block = 256 threads = 8 waves; each wave owns 16 q rows.
// kv streamed in 64-row tiles; both row-major (Ks, for QK^T) and transposed
// (KsT, for PV) copies kept in LDS so every WMMA fragment is a b128 load.
// KX slice is L2-resident (32MB total vs 192MB L2), so single-buffering is fine.
__global__ __launch_bounds__(256) void attn_kernel(const _Float16* __restrict__ QX,
                                                   const _Float16* __restrict__ KX,
                                                   _Float16* __restrict__ O) {
    const int L = 2048, H = 128, NH = 1024;
    const float scale = 0.08838834764831845f;  // 1/sqrt(128)

    __shared__ _Float16 Ks[64][136];     // kv rows x H
    __shared__ _Float16 KsT[128][72];    // H rows x kv  (transposed copy)
    __shared__ _Float16 Ps[8][16][72];   // per-wave probability tile 16x64

    const int t = threadIdx.x, lane = t & 31, wave = t >> 5;
    const int nb = blockIdx.y;
    const int n = nb >> 3, b = nb & 7;           // N=8, B=8
    const int qt = blockIdx.x;
    const int l15 = lane & 15;
    const int kb8 = (lane >> 4) * 8;
    const int kb16 = (lane >> 4) * 16;
    const size_t rowbase = (size_t)b * L;
    const int colbase = n * H;

    // Q fragments for this wave's 16 rows (A operand, K=H=128 -> 4 chunks)
    v16h qf[4];
    {
        const int qrow = qt * 128 + wave * 16 + l15;
        const _Float16* qp = QX + (rowbase + qrow) * NH + colbase;
#pragma unroll
        for (int kc = 0; kc < 4; kc++) {
            v8h lo = *(const v8h*)(qp + kc * 32 + kb8);
            v8h hi = *(const v8h*)(qp + kc * 32 + 16 + kb8);
            qf[kc] = cat8(lo, hi);
        }
    }

    v8f oacc[8];
    float m_i[8], l_i[8];
#pragma unroll
    for (int j = 0; j < 8; j++) {
        m_i[j] = -INFINITY;
        l_i[j] = 0.0f;
#pragma unroll
        for (int e = 0; e < 8; e++) oacc[j][e] = 0.0f;
    }

    const int ldr = t & 63;          // kv row this thread loads
    const int ldh = (t >> 6) * 32;   // h offset (32 halves per thread)

    for (int kv0 = 0; kv0 < L; kv0 += 64) {
        {   // load 64x128 kv tile: row-major into Ks, scattered into KsT
            const _Float16* kp = KX + (rowbase + kv0 + ldr) * NH + colbase + ldh;
            v8h kreg[4];
#pragma unroll
            for (int q = 0; q < 4; q++) kreg[q] = *(const v8h*)(kp + q * 8);
#pragma unroll
            for (int q = 0; q < 4; q++) *(v8h*)&Ks[ldr][ldh + q * 8] = kreg[q];
#pragma unroll
            for (int j = 0; j < 32; j++) KsT[ldh + j][ldr] = kreg[j >> 3][j & 7];
        }
        __syncthreads();

        // scores: s(16 x 64) = q(16x128) * kx^T ; kx row-major == B^T layout
        v8f s[4];
#pragma unroll
        for (int j = 0; j < 4; j++)
#pragma unroll
            for (int e = 0; e < 8; e++) s[j][e] = 0.0f;
#pragma unroll
        for (int kc = 0; kc < 4; kc++)
#pragma unroll
            for (int j = 0; j < 4; j++) {
                const int c = j * 16 + l15;
                v8h lo = *(const v8h*)&Ks[c][kc * 32 + kb8];
                v8h hi = *(const v8h*)&Ks[c][kc * 32 + 16 + kb8];
                s[j] = WMMA16(qf[kc], cat8(lo, hi), s[j]);
            }

        // online softmax: each C-fragment row i lives on 16 lanes sharing (lane>>4)
#pragma unroll
        for (int i = 0; i < 8; i++) {
            float sv0 = s[0][i] * scale, sv1 = s[1][i] * scale;
            float sv2 = s[2][i] * scale, sv3 = s[3][i] * scale;
            float mx = fmaxf(fmaxf(sv0, sv1), fmaxf(sv2, sv3));
#pragma unroll
            for (int off = 1; off < 16; off <<= 1)
                mx = fmaxf(mx, __shfl_xor(mx, off, 32));
            const float mnew = fmaxf(m_i[i], mx);
            const float alpha = __expf(m_i[i] - mnew);
            const float p0 = __expf(sv0 - mnew), p1 = __expf(sv1 - mnew);
            const float p2 = __expf(sv2 - mnew), p3 = __expf(sv3 - mnew);
            const int prow = (lane >> 4) * 8 + i;
            Ps[wave][prow][l15]      = (_Float16)p0;
            Ps[wave][prow][16 + l15] = (_Float16)p1;
            Ps[wave][prow][32 + l15] = (_Float16)p2;
            Ps[wave][prow][48 + l15] = (_Float16)p3;
            float ps = (p0 + p1) + (p2 + p3);
#pragma unroll
            for (int off = 1; off < 16; off <<= 1)
                ps += __shfl_xor(ps, off, 32);
            l_i[i] = l_i[i] * alpha + ps;
            m_i[i] = mnew;
#pragma unroll
            for (int t8 = 0; t8 < 8; t8++) oacc[t8][i] *= alpha;
        }
        // same-wave LDS store->load ordering for Ps
        asm volatile("s_wait_dscnt 0x0" ::: "memory");

        // o += p(16x64) * kx(64x128): A from Ps, B from KsT (both b128 reads)
        v16h pf[2];
#pragma unroll
        for (int kc = 0; kc < 2; kc++) {
            v8h lo = *(const v8h*)&Ps[wave][l15][kc * 32 + kb8];
            v8h hi = *(const v8h*)&Ps[wave][l15][kc * 32 + 16 + kb8];
            pf[kc] = cat8(lo, hi);
        }
#pragma unroll
        for (int t8 = 0; t8 < 8; t8++) {
            const int hc = t8 * 16 + l15;
#pragma unroll
            for (int kc = 0; kc < 2; kc++) {
                v8h lo = *(const v8h*)&KsT[hc][kc * 32 + kb8];
                v8h hi = *(const v8h*)&KsT[hc][kc * 32 + 16 + kb8];
                oacc[t8] = WMMA16(pf[kc], cat8(lo, hi), oacc[t8]);
            }
        }
        __syncthreads();
    }

    float inv[8];
#pragma unroll
    for (int i = 0; i < 8; i++) inv[i] = 1.0f / l_i[i];
    const int qrowb = qt * 128 + wave * 16 + (lane >> 4) * 8;
#pragma unroll
    for (int t8 = 0; t8 < 8; t8++) {
        const int gcol = colbase + t8 * 16 + l15;
#pragma unroll
        for (int i = 0; i < 8; i++)
            O[(rowbase + qrowb + i) * NH + gcol] = (_Float16)(oacc[t8][i] * inv[i]);
    }
}

// ---------------- launcher ----------------
extern "C" void kernel_launch(void* const* d_in, const int* in_sizes, int n_in,
                              void* d_out, int out_size, void* d_ws, size_t ws_size,
                              hipStream_t stream) {
    const float* k   = (const float*)d_in[0];
    const float* q   = (const float*)d_in[1];
    const float* wkx = (const float*)d_in[2];
    const float* wqx = (const float*)d_in[3];
    const float* pw  = (const float*)d_in[4];
    const float* pb  = (const float*)d_in[5];
    float* out = (float*)d_out;

    const int B = 8, L = 2048, E = 1024, N = 8, H = 128;
    const int NH = N * H;        // 1024
    const int M = B * L;         // 16384

    char* ws = (char*)d_ws;
    _Float16* WKt = (_Float16*)(ws);
    _Float16* WQt = (_Float16*)(ws + (size_t)(2) * 1024 * 1024);
    _Float16* Pt  = (_Float16*)(ws + (size_t)(4) * 1024 * 1024);
    _Float16* KXh = (_Float16*)(ws + (size_t)(6) * 1024 * 1024);
    _Float16* QXh = (_Float16*)(ws + (size_t)(38) * 1024 * 1024);
    _Float16* Oh  = (_Float16*)(ws + (size_t)(70) * 1024 * 1024);

    // 1) pack weights to fp16 (transposed for B-operand)
    prep_wt<<<(N * E * H) / 256, 256, 0, stream>>>(wkx, WKt, E, H);
    prep_wt<<<(N * E * H) / 256, 256, 0, stream>>>(wqx, WQt, E, H);
    prep_cast<<<(E * NH) / 256, 256, 0, stream>>>(pw, Pt);

    // 2) head projections as two big GEMMs
    dim3 g1(NH / 128, M / 128);
    gemm128<false, false><<<g1, 256, 0, stream>>>(k, WKt, KXh, nullptr, M, NH, E);
    gemm128<false, false><<<g1, 256, 0, stream>>>(q, WQt, QXh, nullptr, M, NH, E);

    // 3) flash attention per (head, batch)
    dim3 g2(L / 128, N * B);
    attn_kernel<<<g2, 256, 0, stream>>>(QXh, KXh, Oh);

    // 4) output projection + bias (fp32 out)
    dim3 g3(E / 128, M / 128);
    gemm128<true, true><<<g3, 256, 0, stream>>>(Oh, Pt, out, pb, M, E, NH);
}
